// VRNN_59511066853856
// MI455X (gfx1250) — matrix operand order repaired and checked by reading, not compiled, not run on hardware
//
#include <hip/hip_runtime.h>
#include <math.h>

#define B_   64
#define T_   1024
#define DIN  128
#define DH   512
#define DOUT 128

#define SH_STRIDE   520   // 512 + 8 bf16 pad -> 1040B row stride (bank rotate)
#define SHX_STRIDE  136   // 128 + 8 bf16 pad

typedef __attribute__((ext_vector_type(16))) __bf16 v16bf;
typedef __attribute__((ext_vector_type(8)))  __bf16 v8bf;
typedef __attribute__((ext_vector_type(8)))  float  v8f;

// ---------------------------------------------------------------------------
// WMMA fragment loaders (layouts per CDNA5 ISA 7.12.2, wave32)
// A (16x32 bf16, MxK): lanes 0-15 -> row m, K k0+0..7 / k0+16..23
//                      lanes16-31 -> row m, K k0+8..15 / k0+24..31
// B (32x16 bf16, KxN): lane n holds column n; lanes 0-15: K k0+0..7 / k0+16..23,
//                      lanes 16-31: K k0+8..15 / k0+24..31 (same K order as A)
// C/D (16x16 f32):     vgpr r: lanes 0-15 -> (M=r,N=lane); lanes16-31 -> M=r+8
// ---------------------------------------------------------------------------

static __device__ __forceinline__ v16bf load_A_bf16p(const __bf16* X, int stride,
                                                     int lane, int k0) {
  const int m = lane & 15, hi = (lane >> 4) & 1;
  const __bf16* p0 = X + m * stride + k0 + hi * 8;
  v8bf lo = *(const v8bf*)p0;
  v8bf hh = *(const v8bf*)(p0 + 16);
  v16bf a;
#pragma unroll
  for (int i = 0; i < 8; i++) { a[i] = lo[i]; a[8 + i] = hh[i]; }
  return a;
}

// Wt is the N-major (transposed) weight: Wt[n][k], row stride Ktot
static __device__ __forceinline__ v16bf load_B_bf16(const __bf16* Wt, int Ktot,
                                                    int lane, int n0, int k0) {
  const int n = lane & 15, hi = (lane >> 4) & 1;
  const __bf16* p0 = Wt + (long)(n0 + n) * Ktot + k0 + hi * 8;
  v8bf lo = *(const v8bf*)p0;
  v8bf hh = *(const v8bf*)(p0 + 16);
  v16bf b;
#pragma unroll
  for (int i = 0; i < 8; i++) { b[i] = lo[i]; b[8 + i] = hh[i]; }
  return b;
}

static __device__ __forceinline__ v8f wmma_bf16(v16bf a, v16bf b, v8f c) {
  return __builtin_amdgcn_wmma_f32_16x16x32_bf16(false, a, false, b,
                                                 (short)0, c, false, false);
}

// ---------------------------------------------------------------------------
// Weight transpose + fp32->bf16 convert: dst[c][r] = src[r][c]
// ---------------------------------------------------------------------------
__global__ __launch_bounds__(256) void k_convT(const float* __restrict__ src,
                                               __bf16* __restrict__ dst,
                                               int R, int C) {
  int i = blockIdx.x * blockDim.x + threadIdx.x;
  if (i < R * C) {
    int r = i / C, c = i - r * C;
    dst[(long)c * R + r] = (__bf16)src[i];
  }
}

// h0 broadcast into ping buffer 0 (bf16, 64x512) + zero the grid-sync counter
__global__ __launch_bounds__(256) void k_init(const float* __restrict__ h0,
                                              __bf16* __restrict__ hbuf0,
                                              int* __restrict__ sync) {
  int i = blockIdx.x * blockDim.x + threadIdx.x;
  if (i == 0) *sync = 0;
  if (i < B_ * DH) hbuf0[i] = (__bf16)h0[i & (DH - 1)];
}

// ---------------------------------------------------------------------------
// Phase 1: xI = input @ wI into the H region of d_out.
// All 8 waves of a block share one M-strip: stage it (fp32->bf16) in LDS once.
// ---------------------------------------------------------------------------
__global__ __launch_bounds__(256) void k_xproj(const float* __restrict__ in,
                                               const __bf16* __restrict__ wIt,
                                               float* __restrict__ Hbuf) {
  __shared__ __bf16 shA[16 * SHX_STRIDE];           // 4.25 KB
  const int lane = threadIdx.x & 31;
  const int wave = threadIdx.x >> 5;
  const int tile = blockIdx.x * 8 + wave;           // 4096*32 tiles
  const int nt = tile & 31;                         // wave-local N tile
  const int mt = tile >> 5;                         // block-uniform M strip

  // stage 16x128 fp32 strip -> bf16 LDS (2048 elements, 8 per thread)
  {
    const float* strip = in + (long)mt * 16 * DIN;  // rows are contiguous
    const int e = threadIdx.x * 8;
    const int row = e >> 7, col = e & 127;
    v8f x = *(const v8f*)(strip + e);
    v8bf y;
#pragma unroll
    for (int i = 0; i < 8; i++) y[i] = (__bf16)x[i];
    *(v8bf*)&shA[row * SHX_STRIDE + col] = y;
  }
  __syncthreads();

  v8f acc = {0.f, 0.f, 0.f, 0.f, 0.f, 0.f, 0.f, 0.f};
#pragma unroll
  for (int kk = 0; kk < 4; kk++) {
    v16bf a = load_A_bf16p(shA, SHX_STRIDE, lane, kk * 32);
    v16bf b = load_B_bf16(wIt, DIN, lane, nt * 16, kk * 32);
    acc = wmma_bf16(a, b, acc);
  }
  const int col = nt * 16 + (lane & 15);
  const int rb = mt * 16 + ((lane >> 4) << 3);
#pragma unroll
  for (int r = 0; r < 8; r++) Hbuf[(long)(rb + r) * DH + col] = acc[r];
}

// ---------------------------------------------------------------------------
// Phase 2: sequential scan. 8 workgroups x 16 waves = 128 tiles of the
// 64x512 h-state.  wR slice lives in 128 VGPRs per wave for all 1024 steps.
// Per step: h (64x512 bf16, 64KB) is pulled into LDS with async global->LDS
// copies (ASYNCcnt), WMMA A-fragments are fed from LDS (padded rows, no bank
// conflicts), and steps are separated by a device-scope arrival barrier.
// ---------------------------------------------------------------------------
__global__ __launch_bounds__(512) void k_scan(const __bf16* __restrict__ wRt,
                                              float* __restrict__ Hbuf,
                                              __bf16* __restrict__ hping,
                                              int* __restrict__ sync) {
  __shared__ __bf16 sh[B_ * SH_STRIDE];             // 65 KB, padded rows
  const int lane = threadIdx.x & 31;
  const int wave = threadIdx.x >> 5;                // 0..15
  const int tile = blockIdx.x * 16 + wave;          // 0..127
  const int mbase = (tile & 3) * 16;                // M strip
  const int ncol = (tile >> 2) * 16;                // N tile

  // Resident B fragments: wR columns [ncol, ncol+16), all K = 512
  v16bf Breg[16];
#pragma unroll
  for (int kk = 0; kk < 16; kk++)
    Breg[kk] = load_B_bf16(wRt, DH, lane, ncol, kk * 32);

  const int colg = ncol + (lane & 15);
  const int rbase = mbase + ((lane >> 4) << 3);
  __bf16* hb0 = hping;
  __bf16* hb1 = hping + B_ * DH;

  // async-copy chunk assignment: 64 rows x 64 chunks of 16B each
  const unsigned lds_base = (unsigned)(unsigned long)(&sh[0]);

  for (int t = 0; t < T_; t++) {
    const __bf16* hcur = (t & 1) ? hb1 : hb0;
    __bf16* hnxt = (t & 1) ? hb0 : hb1;

    // ---- stage h into LDS: 4096 x 16B chunks, 8 per thread (async) ----
#pragma unroll
    for (int i = 0; i < 8; i++) {
      const int c = i * 512 + threadIdx.x;          // chunk id
      const int row = c >> 6;
      const int off = (c & 63) * 16;                // byte offset in row
      unsigned loff = lds_base + (unsigned)(row * (SH_STRIDE * 2) + off);
      unsigned long gaddr =
          (unsigned long)hcur + (unsigned long)(row * (DH * 2) + off);
      asm volatile("global_load_async_to_lds_b128 %0, %1, off"
                   :: "v"(loff), "v"(gaddr) : "memory");
    }
    asm volatile("s_wait_asynccnt 0x0" ::: "memory");
    __syncthreads();

    // ---- 16 WMMAs: A from LDS, B resident in VGPRs ----
    v8f acc = {0.f, 0.f, 0.f, 0.f, 0.f, 0.f, 0.f, 0.f};
#pragma unroll
    for (int kk = 0; kk < 16; kk++) {
      v16bf a = load_A_bf16p(&sh[mbase * SH_STRIDE], SH_STRIDE, lane, kk * 32);
      acc = wmma_bf16(a, Breg[kk], acc);
    }

    // ---- xI add + tanh + dual store (fp32 H, bf16 next-step operand) ----
#pragma unroll
    for (int r = 0; r < 8; r++) {
      const int row = rbase + r;                          // batch index
      const long idx = ((long)row * T_ + t) * DH + colg;  // H[b][t][c]
      float v = tanhf(Hbuf[idx] + acc[r]);                // xI + h@wR
      Hbuf[idx] = v;
      hnxt[row * DH + colg] = (__bf16)v;
      if (t + 1 < T_) __builtin_prefetch(&Hbuf[idx + DH], 0, 1);
    }

    // ---- device-wide step barrier (8 workgroups) ----
    __threadfence();
    __syncthreads();
    if (threadIdx.x == 0) {
      __hip_atomic_fetch_add(sync, 1, __ATOMIC_RELEASE, __HIP_MEMORY_SCOPE_AGENT);
      while (__hip_atomic_load(sync, __ATOMIC_ACQUIRE, __HIP_MEMORY_SCOPE_AGENT)
             < 8 * (t + 1)) {
        __builtin_amdgcn_s_sleep(8);
      }
    }
    __syncthreads();
  }
}

// ---------------------------------------------------------------------------
// Phase 3: O = H @ wO.  Block-shared M-strip staged fp32->bf16 in LDS once,
// then 8 waves (8 N-tiles) consume it.
// ---------------------------------------------------------------------------
__global__ __launch_bounds__(256) void k_oproj(const float* __restrict__ Hbuf,
                                               const __bf16* __restrict__ wOt,
                                               float* __restrict__ O) {
  __shared__ __bf16 shA[16 * SH_STRIDE];            // 16.25 KB
  const int lane = threadIdx.x & 31;
  const int wave = threadIdx.x >> 5;
  const int tile = blockIdx.x * 8 + wave;           // 4096*8 tiles
  const int nt = tile & 7;                          // wave-local N tile
  const int mt = tile >> 3;                         // block-uniform M strip

  // stage 16x512 fp32 strip -> bf16 LDS (8192 elements, 4 passes x 8/thread)
  {
    const float* strip = Hbuf + (long)mt * 16 * DH; // rows are contiguous
#pragma unroll
    for (int p = 0; p < 4; p++) {
      const int e = p * 2048 + threadIdx.x * 8;
      const int row = e >> 9, col = e & 511;
      v8f x = *(const v8f*)(strip + e);
      v8bf y;
#pragma unroll
      for (int i = 0; i < 8; i++) y[i] = (__bf16)x[i];
      *(v8bf*)&shA[row * SH_STRIDE + col] = y;
    }
  }
  __syncthreads();

  v8f acc = {0.f, 0.f, 0.f, 0.f, 0.f, 0.f, 0.f, 0.f};
#pragma unroll
  for (int kk = 0; kk < 16; kk++) {
    v16bf a = load_A_bf16p(shA, SH_STRIDE, lane, kk * 32);
    v16bf b = load_B_bf16(wOt, DH, lane, nt * 16, kk * 32);
    acc = wmma_bf16(a, b, acc);
  }
  const int col = nt * 16 + (lane & 15);
  const int rb = mt * 16 + ((lane >> 4) << 3);
#pragma unroll
  for (int r = 0; r < 8; r++) O[(long)(rb + r) * DOUT + col] = acc[r];
}

// ---------------------------------------------------------------------------
extern "C" void kernel_launch(void* const* d_in, const int* in_sizes, int n_in,
                              void* d_out, int out_size, void* d_ws,
                              size_t ws_size, hipStream_t stream) {
  const float* in_seq = (const float*)d_in[0];
  const float* h0     = (const float*)d_in[1];
  const float* wI     = (const float*)d_in[2];
  const float* wR     = (const float*)d_in[3];
  const float* wO     = (const float*)d_in[4];

  float* O    = (float*)d_out;                 // [64][1024][128]
  float* Hbuf = O + (long)B_ * T_ * DOUT;      // [64][1024][512]

  // workspace layout (all offsets 128B aligned)
  char* ws = (char*)d_ws;
  __bf16* wIt   = (__bf16*)(ws);                                  // 128 KB
  __bf16* wRt   = (__bf16*)(ws + 131072);                         // 512 KB
  __bf16* wOt   = (__bf16*)(ws + 131072 + 524288);                // 128 KB
  __bf16* hping = (__bf16*)(ws + 131072 + 524288 + 131072);       // 128 KB
  int*    syncc = (int*)   (ws + 131072 + 524288 + 131072 + 131072);

  k_convT<<<(DIN * DH + 255) / 256, 256, 0, stream>>>(wI, wIt, DIN, DH);
  k_convT<<<(DH * DH + 255) / 256, 256, 0, stream>>>(wR, wRt, DH, DH);
  k_convT<<<(DH * DOUT + 255) / 256, 256, 0, stream>>>(wO, wOt, DH, DOUT);
  k_init<<<(B_ * DH + 255) / 256, 256, 0, stream>>>(h0, hping, syncc);

  // xI into H region of d_out: 4096 M-tiles * 32 N-tiles / 8 waves per block
  k_xproj<<<16384, 256, 0, stream>>>(in_seq, wIt, Hbuf);

  // sequential recurrence: 8 co-resident workgroups
  k_scan<<<8, 512, 0, stream>>>(wRt, Hbuf, hping, syncc);

  // output projection: 4096*8 tiles / 8 waves per block
  k_oproj<<<4096, 256, 0, stream>>>(Hbuf, wOt, O);
}
